// Transformer_NoDATA_82781199663417
// MI455X (gfx1250) — matrix operand order repaired and checked by reading, not compiled, not run on hardware
//
#include <hip/hip_runtime.h>
#include <hip/hip_bf16.h>

// ---------------- problem constants ----------------
#define LYR 8
#define DMODEL 1024
#define NHEAD 16
#define HDIM 64
#define FFDIM 4096
#define SEQ 2048
#define BATCH 2
#define MROWS (BATCH * SEQ)   // 4096 tokens
#define QKVW (3 * DMODEL)     // 3072

#define LPITCH 40             // LDS row pitch (bf16 elems): 80B -> conflict-free, 16B aligned

#if defined(__has_builtin)
#if __has_builtin(__builtin_amdgcn_global_load_async_to_lds_b128)
#define HAVE_ASYNC_LDS 1
#endif
#endif
#ifndef HAVE_ASYNC_LDS
#define HAVE_ASYNC_LDS 0
#endif

typedef __attribute__((ext_vector_type(16))) __bf16 v16bf;
typedef __attribute__((ext_vector_type(8)))  __bf16 v8bf;
typedef __attribute__((ext_vector_type(8)))  float  v8f;
typedef __attribute__((ext_vector_type(4)))  int    v4i;

#if HAVE_ASYNC_LDS
typedef v4i __attribute__((address_space(1))) as1_v4i;
typedef v4i __attribute__((address_space(3))) as3_v4i;
#endif

// ---------------- helpers ----------------
__device__ inline __bf16 f2bf(float f) {
  unsigned u = __builtin_bit_cast(unsigned, f);
  u += 0x7FFFu + ((u >> 16) & 1u);   // round-to-nearest-even
  unsigned short h = (unsigned short)(u >> 16);
  return __builtin_bit_cast(__bf16, h);
}

__device__ inline v8f zero8() {
  v8f z = {0.f, 0.f, 0.f, 0.f, 0.f, 0.f, 0.f, 0.f};
  return z;
}

__device__ inline v8f wmma_bf16(v16bf a, v16bf b, v8f c) {
  return __builtin_amdgcn_wmma_f32_16x16x32_bf16(false, a, false, b, (short)0, c,
                                                 false, false);
}

// A fragment (16x32 bf16): 16 contiguous K elements at p.
__device__ inline v16bf load_a_bf(const __bf16* p) {
  v8bf lo = *reinterpret_cast<const v8bf*>(p);
  v8bf hi = *reinterpret_cast<const v8bf*>(p + 8);
  return __builtin_shufflevector(lo, hi, 0, 1, 2, 3, 4, 5, 6, 7, 8, 9, 10, 11,
                                 12, 13, 14, 15);
}
// B fragment (32x16 bf16): 16 contiguous K elements at p.
__device__ inline v16bf load_b_bf(const __bf16* p) {
  v8bf lo = *reinterpret_cast<const v8bf*>(p);
  v8bf hi = *reinterpret_cast<const v8bf*>(p + 8);
  return __builtin_shufflevector(lo, hi, 0, 1, 2, 3, 4, 5, 6, 7, 8, 9, 10, 11,
                                 12, 13, 14, 15);
}

// Copy 8 bf16 (16B) global -> LDS. Async path (ASYNCcnt) when available.
__device__ inline void cp8_g2l(const __bf16* g, __bf16* l) {
#if HAVE_ASYNC_LDS
  __builtin_amdgcn_global_load_async_to_lds_b128((as1_v4i*)g, (as3_v4i*)l, 0, 0);
#else
  *reinterpret_cast<v8bf*>(l) = *reinterpret_cast<const v8bf*>(g);
#endif
}

__device__ inline void wait_async_keep6() {
#if HAVE_ASYNC_LDS
  asm volatile("s_wait_asynccnt 0x6" ::: "memory");
#else
  asm volatile("" ::: "memory");
#endif
}
__device__ inline void wait_async_zero() {
#if HAVE_ASYNC_LDS
  asm volatile("s_wait_asynccnt 0x0" ::: "memory");
#else
  asm volatile("" ::: "memory");
#endif
}

__device__ inline float gelu_exact(float v) {
  return 0.5f * v * (1.0f + erff(v * 0.70710678118654752f));
}

__device__ inline float red16_max(float v) {
#pragma unroll
  for (int m = 1; m <= 8; m <<= 1) v = fmaxf(v, __shfl_xor(v, m, 32));
  return v;
}
__device__ inline float red16_sum(float v) {
#pragma unroll
  for (int m = 1; m <= 8; m <<= 1) v += __shfl_xor(v, m, 32);
  return v;
}

// ---------------- kernels ----------------

// dst(bf16)[0..8n) = src(fp32)
__global__ __launch_bounds__(256) void cvt_kernel(const float* __restrict__ src,
                                                  __bf16* __restrict__ dst,
                                                  int n8) {
  int i = blockIdx.x * 256 + threadIdx.x;
  if (i >= n8) return;
  const float4* s = reinterpret_cast<const float4*>(src) + 2 * (size_t)i;
  float4 f0 = s[0], f1 = s[1];
  v8bf d;
  d[0] = f2bf(f0.x); d[1] = f2bf(f0.y); d[2] = f2bf(f0.z); d[3] = f2bf(f0.w);
  d[4] = f2bf(f1.x); d[5] = f2bf(f1.y); d[6] = f2bf(f1.z); d[7] = f2bf(f1.w);
  reinterpret_cast<v8bf*>(dst)[i] = d;
}

// x = q + pos_emb, dual fp32 + bf16 output.
__global__ __launch_bounds__(256) void add_pos_kernel(const float* __restrict__ q,
                                                      const float* __restrict__ pos,
                                                      float* __restrict__ x,
                                                      __bf16* __restrict__ xb) {
  size_t i = (size_t)blockIdx.x * 256 + threadIdx.x;
  if (i >= (size_t)MROWS * DMODEL) return;
  size_t td = i % ((size_t)SEQ * DMODEL);
  float v = q[i] + pos[td];
  x[i] = v;
  xb[i] = f2bf(v);
}

// C[M,N] = A[M,K](bf16) * W[N,K](bf16)^T with fused epilogue.
// MODE 0: per-col scale; MODE 1: +residual; MODE 2: GELU.
// Block tile 128x256, K-step 32, double-buffered async LDS staging.
// 8 waves (2x4), each 64x64 = 4x4 WMMA fragments fed by ds_load_b128.
template <int MODE, bool WF32, bool WBF16>
__global__ __launch_bounds__(256)
void gemm_kernel(const __bf16* __restrict__ A, const __bf16* __restrict__ W,
                 const float* __restrict__ scale, const float* res,
                 float* C, __bf16* Cb, int Ndim, int Kdim) {
  __shared__ __bf16 Asm[2][128 * LPITCH];
  __shared__ __bf16 Bsm[2][256 * LPITCH];
  const int tid  = threadIdx.x;
  const int lane = tid & 31;
  const int wave = tid >> 5;
  const int mBlk = blockIdx.y * 128;
  const int nBlk = blockIdx.x * 256;
  const int mW   = (wave >> 2) * 64;   // wave tile origin inside block
  const int nW   = (wave & 3) * 64;

  const int r16   = lane & 15;
  const int koffA = (lane >= 16) ? 16 : 0;
  const int koffB = (lane >= 16) ? 16 : 0;

  v8f acc[4][4];
#pragma unroll
  for (int i = 0; i < 4; ++i)
#pragma unroll
    for (int j = 0; j < 4; ++j) acc[i][j] = zero8();

  // cooperative tile staging: A 128x32 (2 chunks/thread), W 256x32 (4 chunks/thread)
  auto stage = [&](int k, int bufi) {
#pragma unroll
    for (int j = 0; j < 2; ++j) {
      int c = tid + j * 256;            // chunk id; 4 chunks of 8 elems per row
      int row = c >> 2, col = (c & 3) * 8;
      cp8_g2l(A + (size_t)(mBlk + row) * Kdim + k + col,
              &Asm[bufi][row * LPITCH + col]);
    }
#pragma unroll
    for (int j = 0; j < 4; ++j) {
      int c = tid + j * 256;
      int row = c >> 2, col = (c & 3) * 8;
      cp8_g2l(W + (size_t)(nBlk + row) * Kdim + k + col,
              &Bsm[bufi][row * LPITCH + col]);
    }
  };

  const int nk = Kdim >> 5;
  stage(0, 0);
  for (int ks = 0; ks < nk; ++ks) {
    const bool more = (ks + 1 < nk);
    if (more) stage((ks + 1) * 32, (ks + 1) & 1);
    if (more) wait_async_keep6(); else wait_async_zero();
    __syncthreads();
    const __bf16* As = &Asm[ks & 1][0];
    const __bf16* Bs = &Bsm[ks & 1][0];
    v16bf a[4], b[4];
#pragma unroll
    for (int mf = 0; mf < 4; ++mf)
      a[mf] = load_a_bf(As + (mW + mf * 16 + r16) * LPITCH + koffA);
#pragma unroll
    for (int nf = 0; nf < 4; ++nf)
      b[nf] = load_b_bf(Bs + (nW + nf * 16 + r16) * LPITCH + koffB);
#pragma unroll
    for (int mf = 0; mf < 4; ++mf)
#pragma unroll
      for (int nf = 0; nf < 4; ++nf)
        acc[mf][nf] = wmma_bf16(a[mf], b[nf], acc[mf][nf]);
    __syncthreads();
  }

  const int hi = (lane >= 16) ? 8 : 0;
#pragma unroll
  for (int mf = 0; mf < 4; ++mf)
#pragma unroll
    for (int nf = 0; nf < 4; ++nf)
#pragma unroll
      for (int r = 0; r < 8; ++r) {
        int row = mBlk + mW + mf * 16 + r + hi;
        int col = nBlk + nW + nf * 16 + r16;
        float v = acc[mf][nf][r];
        if (MODE == 0) v *= scale[col];
        if (MODE == 2) v = gelu_exact(v);
        if (MODE == 1) v += res[(size_t)row * Ndim + col];
        if (WF32)  C[(size_t)row * Ndim + col] = v;
        if (WBF16) Cb[(size_t)row * Ndim + col] = f2bf(v);
      }
}

// L2-normalize q,k head vectors (fp32 in) and emit full bf16 qkv copy.
__global__ __launch_bounds__(256)
void l2norm_cvt_kernel(const float* __restrict__ qkv, __bf16* __restrict__ qkvb) {
  const int lane = threadIdx.x & 31;
  const int wave = threadIdx.x >> 5;
  const int idx  = blockIdx.x * 8 + wave;   // B*T*H
  const int h = idx & 15;
  const int t = (idx >> 4) & (SEQ - 1);
  const int b = idx >> 15;
  const size_t base = ((size_t)b * SEQ + t) * QKVW + h * HDIM;
#pragma unroll
  for (int part = 0; part < 2; ++part) {    // q then k: normalize
    const float* p = qkv + base + part * DMODEL;
    __bf16* pb = qkvb + base + part * DMODEL;
    float x0 = p[lane], x1 = p[lane + 32];
    float ss = x0 * x0 + x1 * x1;
#pragma unroll
    for (int m = 1; m <= 16; m <<= 1) ss += __shfl_xor(ss, m, 32);
    float inv = 1.0f / fmaxf(sqrtf(ss), 1e-5f);
    pb[lane] = f2bf(x0 * inv);
    pb[lane + 32] = f2bf(x1 * inv);
  }
  {                                          // v: plain convert
    const float* p = qkv + base + 2 * DMODEL;
    __bf16* pb = qkvb + base + 2 * DMODEL;
    pb[lane] = f2bf(p[lane]);
    pb[lane + 32] = f2bf(p[lane + 32]);
  }
}

// Causal flash attention, bf16 operands, fp32 accumulation.
__global__ __launch_bounds__(256)
void attn_kernel(const __bf16* __restrict__ qkv, __bf16* __restrict__ y) {
  __shared__ __bf16 plds_all[8][16 * 32];
  const int lane = threadIdx.x & 31;
  const int wave = threadIdx.x >> 5;
  __bf16* plds = &plds_all[wave][0];

  const int tile = blockIdx.x * 8 + wave;        // B*H*(T/16) tiles
  const int qt = tile & (SEQ / 16 - 1);
  const int h  = (tile >> 7) & (NHEAD - 1);
  const int b  = tile >> 11;
  const int qBase = qt * 16;

  const size_t strideT = QKVW;
  const __bf16* qptr = qkv + ((size_t)b * SEQ) * strideT + h * HDIM;
  const __bf16* kptr = qptr + DMODEL;
  const __bf16* vptr = qptr + 2 * DMODEL;

  const int r16   = lane & 15;
  const int hi    = (lane >= 16) ? 8 : 0;
  const int koffA = (lane >= 16) ? 16 : 0;
  const int koffB = (lane >= 16) ? 16 : 0;

  v16bf qa[2];
#pragma unroll
  for (int f = 0; f < 2; ++f)
    qa[f] = load_a_bf(qptr + (size_t)(qBase + r16) * strideT + f * 32 + koffA);

  v8f o[4];
  float mrow[8], lrow[8];
#pragma unroll
  for (int dt = 0; dt < 4; ++dt) o[dt] = zero8();
#pragma unroll
  for (int r = 0; r < 8; ++r) { mrow[r] = -1e30f; lrow[r] = 0.f; }

  const float sc = 0.125f;  // 1/sqrt(64)
  for (int kt = 0; kt <= qBase + 15; kt += 32) {
    v8f s0 = zero8(), s1 = zero8();
#pragma unroll
    for (int ks = 0; ks < 2; ++ks) {
      v16bf kb0 = load_b_bf(kptr + (size_t)(kt + r16) * strideT + ks * 32 + koffB);
      s0 = wmma_bf16(qa[ks], kb0, s0);
      v16bf kb1 =
          load_b_bf(kptr + (size_t)(kt + 16 + r16) * strideT + ks * 32 + koffB);
      s1 = wmma_bf16(qa[ks], kb1, s1);
    }
#pragma unroll
    for (int r = 0; r < 8; ++r) {
      const int qrow = qBase + r + hi;
      float a0 = s0[r] * sc, a1 = s1[r] * sc;
      if (kt + r16 > qrow)      a0 = -1e30f;   // causal mask
      if (kt + 16 + r16 > qrow) a1 = -1e30f;
      float rmax = red16_max(fmaxf(a0, a1));
      float newm = fmaxf(mrow[r], rmax);
      float resc = __expf(mrow[r] - newm);
      mrow[r] = newm;
      float e0 = __expf(a0 - newm);
      float e1 = __expf(a1 - newm);
      lrow[r] = lrow[r] * resc + red16_sum(e0 + e1);
#pragma unroll
      for (int dt = 0; dt < 4; ++dt) o[dt][r] *= resc;
      plds[(r + hi) * 32 + r16]      = f2bf(e0);
      plds[(r + hi) * 32 + 16 + r16] = f2bf(e1);
    }
    asm volatile("s_wait_dscnt 0x0" ::: "memory");
    v16bf pa = load_a_bf(plds + r16 * 32 + koffA);
    const int kk = kt + ((lane >= 16) ? 16 : 0);
#pragma unroll
    for (int dt = 0; dt < 4; ++dt) {
      const __bf16* vp = vptr + (size_t)kk * strideT + dt * 16 + r16;
      v16bf vb;
#pragma unroll
      for (int j = 0; j < 16; ++j) vb[j] = vp[(size_t)j * strideT];
      o[dt] = wmma_bf16(pa, vb, o[dt]);
    }
  }

#pragma unroll
  for (int dt = 0; dt < 4; ++dt)
#pragma unroll
    for (int r = 0; r < 8; ++r) {
      const int q = qBase + r + hi;
      y[((size_t)b * SEQ + q) * DMODEL + h * HDIM + dt * 16 + r16] =
          f2bf(o[dt][r] / lrow[r]);
    }
}

// Final LayerNorm * gamma. One block per token.
__global__ __launch_bounds__(256)
void layernorm_kernel(const float* __restrict__ x, const float* __restrict__ g,
                      float* __restrict__ out) {
  __shared__ float sred[16];
  const float* xp = x + (size_t)blockIdx.x * DMODEL;
  float s = 0.f, s2 = 0.f;
  for (int i = threadIdx.x; i < DMODEL; i += 256) {
    float v = xp[i];
    s += v; s2 += v * v;
  }
#pragma unroll
  for (int m = 1; m <= 16; m <<= 1) {
    s += __shfl_xor(s, m, 32);
    s2 += __shfl_xor(s2, m, 32);
  }
  const int wv = threadIdx.x >> 5, ln = threadIdx.x & 31;
  if (ln == 0) { sred[wv] = s; sred[8 + wv] = s2; }
  __syncthreads();
  s = 0.f; s2 = 0.f;
#pragma unroll
  for (int w = 0; w < 8; ++w) { s += sred[w]; s2 += sred[8 + w]; }
  const float mu  = s * (1.0f / DMODEL);
  const float var = s2 * (1.0f / DMODEL) - mu * mu;
  const float inv = rsqrtf(var + 1e-5f);
  for (int i = threadIdx.x; i < DMODEL; i += 256)
    out[(size_t)blockIdx.x * DMODEL + i] = (xp[i] - mu) * inv * g[i];
}

// ---------------- host launcher ----------------
static inline void cvt(const float* s, __bf16* d, size_t n, hipStream_t st) {
  int n8 = (int)(n / 8);
  cvt_kernel<<<(n8 + 255) / 256, 256, 0, st>>>(s, d, n8);
}

extern "C" void kernel_launch(void* const* d_in, const int* in_sizes, int n_in,
                              void* d_out, int out_size, void* d_ws, size_t ws_size,
                              hipStream_t stream) {
  (void)in_sizes; (void)n_in; (void)out_size; (void)ws_size;
  const float* q     = (const float*)d_in[0];
  const float* pos   = (const float*)d_in[1];
  const float* Wqkv  = (const float*)d_in[2];
  const float* convw = (const float*)d_in[3];
  const float* projw = (const float*)d_in[4];
  const float* fcw   = (const float*)d_in[5];
  const float* mlpw  = (const float*)d_in[6];
  const float* gamma = (const float*)d_in[7];
  float* out = (float*)d_out;

  // fp32 buffers
  float* x    = (float*)d_ws;                        // M*D
  float* qkvf = x + (size_t)MROWS * DMODEL;          // M*3D
  // bf16 buffers
  __bf16* xb   = (__bf16*)(qkvf + (size_t)MROWS * QKVW);  // M*D
  __bf16* qkvb = xb + (size_t)MROWS * DMODEL;             // M*3D
  __bf16* ybuf = qkvb + (size_t)MROWS * QKVW;             // M*D
  __bf16* hb   = ybuf + (size_t)MROWS * DMODEL;           // M*FF
  __bf16* wb   = hb + (size_t)MROWS * FFDIM;              // FF*D weight scratch

  add_pos_kernel<<<(MROWS * DMODEL + 255) / 256, 256, 0, stream>>>(q, pos, x, xb);

  for (int l = 0; l < LYR; ++l) {
    const float* Wq = Wqkv + (size_t)l * QKVW * DMODEL;
    const float* cv = convw + (size_t)l * QKVW;
    const float* Wp = projw + (size_t)l * DMODEL * DMODEL;
    const float* Wf = fcw + (size_t)l * FFDIM * DMODEL;
    const float* Wm = mlpw + (size_t)l * DMODEL * FFDIM;

    cvt(Wq, wb, (size_t)QKVW * DMODEL, stream);
    gemm_kernel<0, true, false><<<dim3(QKVW / 256, MROWS / 128), 256, 0, stream>>>(
        xb, wb, cv, nullptr, qkvf, nullptr, QKVW, DMODEL);

    l2norm_cvt_kernel<<<(MROWS * NHEAD) / 8, 256, 0, stream>>>(qkvf, qkvb);
    attn_kernel<<<(BATCH * NHEAD * (SEQ / 16)) / 8, 256, 0, stream>>>(qkvb, ybuf);

    cvt(Wp, wb, (size_t)DMODEL * DMODEL, stream);
    gemm_kernel<1, true, true><<<dim3(DMODEL / 256, MROWS / 128), 256, 0, stream>>>(
        ybuf, wb, nullptr, x, x, xb, DMODEL, DMODEL);

    cvt(Wf, wb, (size_t)FFDIM * DMODEL, stream);
    gemm_kernel<2, false, true><<<dim3(FFDIM / 256, MROWS / 128), 256, 0, stream>>>(
        xb, wb, nullptr, nullptr, nullptr, hb, FFDIM, DMODEL);

    cvt(Wm, wb, (size_t)DMODEL * FFDIM, stream);
    gemm_kernel<1, true, true><<<dim3(DMODEL / 256, MROWS / 128), 256, 0, stream>>>(
        hb, wb, nullptr, x, x, xb, DMODEL, FFDIM);
  }

  layernorm_kernel<<<MROWS, 256, 0, stream>>>(x, gamma, out);
}
